// GraphNetMultiCls_86011015070504
// MI455X (gfx1250) — compile-verified, matrix operand order checked by reading, not run on hardware
//
#include <hip/hip_runtime.h>
#include <math.h>

// ---------------------------------------------------------------------------
// Types for CDNA5 WMMA (gfx1250, wave32)
// ---------------------------------------------------------------------------
typedef __attribute__((ext_vector_type(16))) __bf16 v16bf;
typedef __attribute__((ext_vector_type(8)))  float  v8f;

#define DEV __device__ __forceinline__

static constexpr int NN  = 100000;
static constexpr int EE  = 500000;
static constexpr int FIN = 512;
static constexpr int NH  = 128;
static constexpr int KK1 = 20000;   // ceil(0.2*N)
static constexpr int KK2 = 4000;
static constexpr int KK3 = 800;

// float -> bf16, round-to-nearest-even
DEV unsigned short f2bf(float f){
  unsigned u = __float_as_uint(f);
  u += 0x7FFFu + ((u >> 16) & 1u);
  return (unsigned short)(u >> 16);
}
// monotone order-preserving float->uint key (for radix select of largest)
DEV unsigned f2ord(float f){
  unsigned u = __float_as_uint(f);
  return (u & 0x80000000u) ? ~u : (u | 0x80000000u);
}
// K index inside a 16-bit 16x32 A fragment: element j (0..15), lane-half hh
// (ISA 7.12.2: VGPR0 holds K=0,1 (lanes 0-15) / K=8,9 (lanes 16-31), ...,
//  VGPR4..7 hold K=16.. / K=24..)
DEV int klocal(int j, int hh){
  int v = j >> 1, pb = j & 1;
  return ((v & 3) << 1) + ((v >> 2) << 4) + (hh << 3) + pb;
}

// ---------------------------------------------------------------------------
// Utility fills
// ---------------------------------------------------------------------------
__global__ void k_fill_f32(float* p, float v, long n){
  for(long i = (long)blockIdx.x*blockDim.x + threadIdx.x; i < n;
      i += (long)gridDim.x*blockDim.x) p[i] = v;
}
__global__ void k_fill_i32(int* p, int v, long n){
  for(long i = (long)blockIdx.x*blockDim.x + threadIdx.x; i < n;
      i += (long)gridDim.x*blockDim.x) p[i] = v;
}

__global__ void k_init_edges(const int* ei, int* src, int* dst, float* ev, int E){
  int e = blockIdx.x*blockDim.x + threadIdx.x;
  if(e >= E) return;
  src[e] = ei[e];
  dst[e] = ei[E + e];
  ev[e]  = 1.0f;
}

// ---------------------------------------------------------------------------
// bf16 fragment packing (A-matrix layout for features, B-matrix for weights)
// ---------------------------------------------------------------------------
// packed feature layout: frag_idx = ((rb*T + t)*32 + lane)*16 + j,
//   node = rb*16 + (lane&15), f = t*32 + klocal(j, lane>>4), T = F/32
__global__ void k_pack_feat(const float* x, unsigned short* xp, int n, int F, int tshift){
  long total = (long)n * F;
  int T = F >> 5;
  for(long idx = (long)blockIdx.x*blockDim.x + threadIdx.x; idx < total;
      idx += (long)gridDim.x*blockDim.x){
    int  j    = (int)(idx & 15);
    int  lane = (int)((idx >> 4) & 31);
    long rest = idx >> 9;
    int  t    = (int)(rest & (T - 1));
    long rb   = rest >> tshift;
    int  node = (int)((rb << 4) + (lane & 15));
    int  f    = (t << 5) + klocal(j, lane >> 4);
    float v   = (node < n) ? x[(size_t)node*F + f] : 0.0f;
    xp[idx]   = f2bf(v);
  }
}
// same, but value = agg/max(cnt,1)  (mean aggregation finalize)
__global__ void k_mean_pack(const float* agg, const float* cnt, unsigned short* mp,
                            int n, int F, int tshift){
  long total = (long)n * F;
  int T = F >> 5;
  for(long idx = (long)blockIdx.x*blockDim.x + threadIdx.x; idx < total;
      idx += (long)gridDim.x*blockDim.x){
    int  j    = (int)(idx & 15);
    int  lane = (int)((idx >> 4) & 31);
    long rest = idx >> 9;
    int  t    = (int)(rest & (T - 1));
    long rb   = rest >> tshift;
    int  node = (int)((rb << 4) + (lane & 15));
    float v = 0.0f;
    if(node < n){
      int f = (t << 5) + klocal(j, lane >> 4);
      v = agg[(size_t)node*F + f] / fmaxf(cnt[node], 1.0f);
    }
    mp[idx] = f2bf(v);
  }
}
// weight [F,128] row-major -> B fragments: frag_idx = ((t*8 + nt)*32 + lane)*16 + j
//   k = t*32 + klocal(j, lane>>4), col = nt*16 + (lane&15)
__global__ void k_pack_w(const float* w, unsigned short* wp, int F){
  int total = F * 128;
  for(int idx = blockIdx.x*blockDim.x + threadIdx.x; idx < total;
      idx += gridDim.x*blockDim.x){
    int j    = idx & 15;
    int lane = (idx >> 4) & 31;
    int rest = idx >> 9;
    int nt   = rest & 7;
    int t    = rest >> 3;
    int k    = (t << 5) + klocal(j, lane >> 4);
    int col  = (nt << 4) + (lane & 15);
    wp[idx]  = f2bf(w[(size_t)k*128 + col]);
  }
}

// ---------------------------------------------------------------------------
// Edge scatter: agg[dst] += x[src]*ev ; cnt[dst] += ev
// ---------------------------------------------------------------------------
__global__ void k_scatter(const float* __restrict__ x, const int* src, const int* dst,
                          const float* ev, float* agg, float* cnt, int E, int cshift){
  int C = 1 << cshift;                 // F/4 chunks of 4 floats
  long total = (long)E << cshift;
  for(long i = (long)blockIdx.x*blockDim.x + threadIdx.x; i < total;
      i += (long)gridDim.x*blockDim.x){
    int e = (int)(i >> cshift);
    int c = (int)(i & (C - 1));
    float w = ev[e];
    if(w == 0.0f) continue;
    int s = src[e], d = dst[e];
    const float* xs = x + (size_t)s*(C << 2) + (c << 2);
    float*       ad = agg + (size_t)d*(C << 2) + (c << 2);
    __builtin_prefetch(xs, 0, 1);      // global_prefetch_b8
    #pragma unroll
    for(int q = 0; q < 4; ++q) atomicAdd(ad + q, xs[q]*w);
    if(c == 0) atomicAdd(cnt + d, w);
  }
}

// ---------------------------------------------------------------------------
// SAGEConv GEMM: h = relu(mean @ Wl + b + x @ Wr), all bf16 operands, f32 acc.
// One wave per 16x16 tile; 8 waves/block cover the 128 output columns.
// ---------------------------------------------------------------------------
__global__ void k_gemm_sage(const unsigned short* __restrict__ meanP,
                            const unsigned short* __restrict__ xP,
                            const unsigned short* __restrict__ wlP,
                            const unsigned short* __restrict__ wrP,
                            const float* __restrict__ bias,
                            float* __restrict__ h, int n, int F){
  int lane = threadIdx.x & 31;
  int wv   = threadIdx.x >> 5;          // 0..7: output column tile
  int rb   = blockIdx.x;                // 16-row tile
  int T    = F >> 5;                    // K steps of 32
  const v16bf* amB = (const v16bf*)meanP + (size_t)rb*T*32 + lane;
  const v16bf* axB = (const v16bf*)xP    + (size_t)rb*T*32 + lane;
  const v16bf* wlB = (const v16bf*)wlP   + wv*32 + lane;
  const v16bf* wrB = (const v16bf*)wrP   + wv*32 + lane;
  v8f acc = {};
  for(int t = 0; t < T; ++t){
    v16bf aM = amB[(size_t)t*32];
    v16bf aX = axB[(size_t)t*32];
    v16bf bL = wlB[(size_t)t*256];
    v16bf bR = wrB[(size_t)t*256];
    acc = __builtin_amdgcn_wmma_f32_16x16x32_bf16(false, aM, false, bL,
                                                  (short)0, acc, false, false);
    acc = __builtin_amdgcn_wmma_f32_16x16x32_bf16(false, aX, false, bR,
                                                  (short)0, acc, false, false);
  }
  int col = (wv << 4) | (lane & 15);
  float bv = bias[col];
  int rbase = (rb << 4) + ((lane >> 4) << 3);   // C layout: VGPR i -> M=i (+8 hi half)
  #pragma unroll
  for(int i = 0; i < 8; ++i){
    int row = rbase + i;
    if(row < n){
      float v = acc[i] + bv;
      h[(size_t)row*128 + col] = v > 0.0f ? v : 0.0f;
    }
  }
}

// ---------------------------------------------------------------------------
// GCN score pieces
// ---------------------------------------------------------------------------
__global__ void k_xw(const float* __restrict__ h, const float* __restrict__ gw,
                     float* xw, int n){
  int warp = (blockIdx.x*blockDim.x + threadIdx.x) >> 5;
  int lane = threadIdx.x & 31;
  if(warp >= n) return;
  const float* row = h + (size_t)warp*128;
  float s = 0.0f;
  for(int c = lane; c < 128; c += 32) s += row[c]*gw[c];
  for(int off = 16; off; off >>= 1) s += __shfl_down(s, off, 32);
  if(lane == 0) xw[warp] = s;
}
__global__ void k_deg(const int* dst, const float* ev, float* deg, int E){
  int e = blockIdx.x*blockDim.x + threadIdx.x;
  if(e >= E) return;
  float w = ev[e];
  if(w != 0.0f) atomicAdd(deg + dst[e], w);
}
__global__ void k_edge_score(const int* src, const int* dst, const float* ev,
                             const float* xw, const float* deg, float* sagg, int E){
  int e = blockIdx.x*blockDim.x + threadIdx.x;
  if(e >= E) return;
  float w = ev[e];
  if(w == 0.0f) return;
  int s = src[e], d = dst[e];
  float norm = rsqrtf(deg[s])*rsqrtf(deg[d])*w;
  atomicAdd(sagg + d, xw[s]*norm);
}
__global__ void k_score_final(const float* sagg, const float* xw, const float* deg,
                              const float* gb, float* score, unsigned* keys, int n){
  int i = blockIdx.x*blockDim.x + threadIdx.x;
  if(i >= n) return;
  float sc = sagg[i] + xw[i]/deg[i] + gb[0];
  score[i] = sc;
  keys[i]  = f2ord(sc);
}

// ---------------------------------------------------------------------------
// Exact top-k: radix select of k-th largest key + compaction with tie handling
// ---------------------------------------------------------------------------
__global__ void k_radix_select(const unsigned* __restrict__ keys, int n, int k,
                               unsigned* out){
  __shared__ unsigned hist[256];
  __shared__ unsigned sel_prefix;
  __shared__ int      sel_rem;
  if(threadIdx.x == 0){ sel_prefix = 0; sel_rem = k; }
  __syncthreads();
  for(int shift = 24; shift >= 0; shift -= 8){
    hist[threadIdx.x] = 0;
    __syncthreads();
    unsigned pmask = (shift == 24) ? 0u : (0xFFFFFFFFu << (shift + 8));
    unsigned pref  = sel_prefix;
    for(int i = threadIdx.x; i < n; i += blockDim.x){
      unsigned key = keys[i];
      if((key & pmask) == pref) atomicAdd(&hist[(key >> shift) & 255], 1u);
    }
    __syncthreads();
    if(threadIdx.x == 0){
      int rem = sel_rem;
      int b = 255;
      for(;;){
        int c = (int)hist[b];
        if(c >= rem) break;
        rem -= c;
        if(b == 0) break;
        --b;
      }
      sel_prefix = pref | ((unsigned)b << shift);
      sel_rem    = rem;
    }
    __syncthreads();
  }
  if(threadIdx.x == 0) out[0] = sel_prefix;
}
__global__ void k_count_greater(const unsigned* keys, int n, const unsigned* thr,
                                int* cntG){
  int i = blockIdx.x*blockDim.x + threadIdx.x;
  if(i >= n) return;
  if(keys[i] > thr[0]) atomicAdd(cntG, 1);
}
__global__ void k_compact(const unsigned* keys, const unsigned* thr, const int* cntG,
                          int* counters, int* perm, int n, int k){
  int i = blockIdx.x*blockDim.x + threadIdx.x;
  if(i >= n) return;
  unsigned T = thr[0], key = keys[i];
  if(key > T){
    int pos = atomicAdd(&counters[0], 1);
    perm[pos] = i;
  } else if(key == T){
    int pos = cntG[0] + atomicAdd(&counters[1], 1);
    if(pos < k) perm[pos] = i;
  }
}
__global__ void k_scatter_rank(int* rank, const int* perm, int k){
  int i = blockIdx.x*blockDim.x + threadIdx.x;
  if(i >= k) return;
  rank[perm[i]] = i;
}
__global__ void k_pool(const float* h, const float* score, const int* perm,
                       float* xk, int k){
  long total = (long)k*128;
  long i = (long)blockIdx.x*blockDim.x + threadIdx.x;
  if(i >= total) return;
  int r = (int)(i >> 7), c = (int)(i & 127);
  int p = perm[r];
  xk[i] = h[(size_t)p*128 + c] * tanhf(score[p]);
}
__global__ void k_relabel(int* src, int* dst, float* ev, const int* rank, int E){
  int e = blockIdx.x*blockDim.x + threadIdx.x;
  if(e >= E) return;
  int ns = rank[src[e]], nd = rank[dst[e]];
  float w = ev[e];
  ev[e]  = (ns >= 0 && nd >= 0) ? w : 0.0f;
  src[e] = ns < 0 ? 0 : ns;
  dst[e] = nd < 0 ? 0 : nd;
}
__global__ void k_readout(const float* xk, int k, float* xo){
  int c = blockIdx.x;               // 0..127
  __shared__ float smax[256], ssum[256];
  float m = -INFINITY, s = 0.0f;
  for(int r = threadIdx.x; r < k; r += blockDim.x){
    float v = xk[(size_t)r*128 + c];
    m = fmaxf(m, v);
    s += v;
  }
  smax[threadIdx.x] = m; ssum[threadIdx.x] = s;
  __syncthreads();
  for(int off = 128; off; off >>= 1){
    if((int)threadIdx.x < off){
      smax[threadIdx.x] = fmaxf(smax[threadIdx.x], smax[threadIdx.x + off]);
      ssum[threadIdx.x] += ssum[threadIdx.x + off];
    }
    __syncthreads();
  }
  if(threadIdx.x == 0){ xo[c] = smax[0]; xo[128 + c] = ssum[0]/(float)k; }
}

// ---------------------------------------------------------------------------
// Final MLP: g=x1+x2+x3; relu(g@lw1+lb1); features=relu(@lw2+lb2); out=@lw3+lb3
// ---------------------------------------------------------------------------
__global__ void k_mlp(const float* x1, const float* x2, const float* x3,
                      const float* lw1, const float* lb1,
                      const float* lw2, const float* lb2,
                      const float* lw3, const float* lb3, float* dout){
  __shared__ float g[256], t1[128], t2[32];
  int t = threadIdx.x;
  if(t < 256) g[t] = x1[t] + x2[t] + x3[t];
  __syncthreads();
  if(t < 128){
    float s = lb1[t];
    for(int i = 0; i < 256; ++i) s += g[i]*lw1[i*128 + t];
    t1[t] = s > 0.0f ? s : 0.0f;
  }
  __syncthreads();
  if(t < 32){
    float s = lb2[t];
    for(int i = 0; i < 128; ++i) s += t1[i]*lw2[i*32 + t];
    float f = s > 0.0f ? s : 0.0f;
    t2[t] = f;
    dout[t] = f;                        // features [32]
  }
  __syncthreads();
  if(t < 3){
    float s = lb3[t];
    for(int i = 0; i < 32; ++i) s += t2[i]*lw3[i*3 + t];
    dout[32 + t] = s;                   // logits [3]
  }
}

// ---------------------------------------------------------------------------
// Host orchestration
// ---------------------------------------------------------------------------
static inline unsigned gsz(long total){
  long b = (total + 255) / 256;
  if(b < 1) b = 1;
  if(b > 1048576) b = 1048576;
  return (unsigned)b;
}

extern "C" void kernel_launch(void* const* d_in, const int* in_sizes, int n_in,
                              void* d_out, int out_size, void* d_ws, size_t ws_size,
                              hipStream_t stream){
  (void)in_sizes; (void)n_in; (void)out_size; (void)ws_size;
  const float* x   = (const float*)d_in[0];
  const int*   ei  = (const int*)  d_in[1];
  const float* w1l = (const float*)d_in[2];
  const float* b1l = (const float*)d_in[3];
  const float* w1r = (const float*)d_in[4];
  const float* g1w = (const float*)d_in[5];
  const float* g1b = (const float*)d_in[6];
  const float* w2l = (const float*)d_in[7];
  const float* b2l = (const float*)d_in[8];
  const float* w2r = (const float*)d_in[9];
  const float* g2w = (const float*)d_in[10];
  const float* g2b = (const float*)d_in[11];
  const float* w3l = (const float*)d_in[12];
  const float* b3l = (const float*)d_in[13];
  const float* w3r = (const float*)d_in[14];
  const float* g3w = (const float*)d_in[15];
  const float* g3b = (const float*)d_in[16];
  const float* lw1 = (const float*)d_in[17];
  const float* lb1 = (const float*)d_in[18];
  const float* lw2 = (const float*)d_in[19];
  const float* lb2 = (const float*)d_in[20];
  const float* lw3 = (const float*)d_in[21];
  const float* lb3 = (const float*)d_in[22];

  // ---- workspace carve-up ----
  char* wp = (char*)d_ws;
  auto take = [&](size_t bytes) -> void* {
    void* r = (void*)wp;
    wp += (bytes + 255) & ~(size_t)255;
    return r;
  };
  float*          agg   = (float*)take((size_t)NN*FIN*4);
  float*          cnt   = (float*)take((size_t)NN*4);
  float*          deg   = (float*)take((size_t)NN*4);
  float*          xwv   = (float*)take((size_t)NN*4);
  float*          sagg  = (float*)take((size_t)NN*4);
  float*          score = (float*)take((size_t)NN*4);
  unsigned*       keys  = (unsigned*)take((size_t)NN*4);
  int*            rank  = (int*)take((size_t)NN*4);
  int*            perm  = (int*)take((size_t)KK1*4);
  int*            srcA  = (int*)take((size_t)EE*4);
  int*            dstA  = (int*)take((size_t)EE*4);
  float*          evA   = (float*)take((size_t)EE*4);
  unsigned short* meanP = (unsigned short*)take((size_t)NN*FIN*2);
  unsigned short* xP    = (unsigned short*)take((size_t)NN*FIN*2);
  unsigned short* wlP   = (unsigned short*)take((size_t)FIN*NH*2);
  unsigned short* wrP   = (unsigned short*)take((size_t)FIN*NH*2);
  float*          h     = (float*)take((size_t)NN*NH*4);
  float*          xk    = (float*)take((size_t)KK1*NH*4);
  float*          x1v   = (float*)take(256*4);
  float*          x2v   = (float*)take(256*4);
  float*          x3v   = (float*)take(256*4);
  unsigned*       thr   = (unsigned*)take(16);
  int*            ctr   = (int*)take(16);   // [0],[1]=compaction counters, [2]=cntGreater

  k_init_edges<<<gsz(EE), 256, 0, stream>>>(ei, srcA, dstA, evA, EE);

  auto run_stage = [&](int n, int F, int kout, const float* curx,
                       const float* wl, const float* wr, const float* bias,
                       const float* gw, const float* gb, float* xout){
    long nF = (long)n*F;
    int tshift = (F == 512) ? 4 : 2;    // log2(F/32)
    int cshift = (F == 512) ? 7 : 5;    // log2(F/4)
    k_fill_f32<<<gsz(nF), 256, 0, stream>>>(agg, 0.0f, nF);
    k_fill_f32<<<gsz(n),  256, 0, stream>>>(cnt, 0.0f, n);
    k_pack_w   <<<gsz((long)F*128), 256, 0, stream>>>(wl, wlP, F);
    k_pack_w   <<<gsz((long)F*128), 256, 0, stream>>>(wr, wrP, F);
    k_pack_feat<<<gsz(nF), 256, 0, stream>>>(curx, xP, n, F, tshift);
    k_scatter  <<<4096, 256, 0, stream>>>(curx, srcA, dstA, evA, agg, cnt, EE, cshift);
    k_mean_pack<<<gsz(nF), 256, 0, stream>>>(agg, cnt, meanP, n, F, tshift);
    k_gemm_sage<<<n/16, 256, 0, stream>>>(meanP, xP, wlP, wrP, bias, h, n, F);
    // GCN score
    k_fill_f32<<<gsz(n), 256, 0, stream>>>(deg,  1.0f, n);   // self-loop
    k_fill_f32<<<gsz(n), 256, 0, stream>>>(sagg, 0.0f, n);
    k_deg       <<<gsz(EE), 256, 0, stream>>>(dstA, evA, deg, EE);
    k_xw        <<<gsz((long)n*32), 256, 0, stream>>>(h, gw, xwv, n);
    k_edge_score<<<gsz(EE), 256, 0, stream>>>(srcA, dstA, evA, xwv, deg, sagg, EE);
    k_score_final<<<gsz(n), 256, 0, stream>>>(sagg, xwv, deg, gb, score, keys, n);
    // exact top-k
    k_fill_i32<<<1, 64, 0, stream>>>(ctr, 0, 3);
    k_radix_select <<<1, 256, 0, stream>>>(keys, n, kout, thr);
    k_count_greater<<<gsz(n), 256, 0, stream>>>(keys, n, thr, ctr + 2);
    k_compact      <<<gsz(n), 256, 0, stream>>>(keys, thr, ctr + 2, ctr, perm, n, kout);
    k_fill_i32     <<<gsz(n), 256, 0, stream>>>(rank, -1, n);
    k_scatter_rank <<<gsz(kout), 256, 0, stream>>>(rank, perm, kout);
    k_pool   <<<gsz((long)kout*128), 256, 0, stream>>>(h, score, perm, xk, kout);
    k_relabel<<<gsz(EE), 256, 0, stream>>>(srcA, dstA, evA, rank, EE);
    k_readout<<<128, 256, 0, stream>>>(xk, kout, xout);
  };

  run_stage(NN,  FIN, KK1, x,  w1l, w1r, b1l, g1w, g1b, x1v);
  run_stage(KK1, NH,  KK2, xk, w2l, w2r, b2l, g2w, g2b, x2v);
  run_stage(KK2, NH,  KK3, xk, w3l, w3r, b3l, g3w, g3b, x3v);

  k_mlp<<<1, 256, 0, stream>>>(x1v, x2v, x3v, lw1, lb1, lw2, lb2, lw3, lb3,
                               (float*)d_out);
}
